// GQA_47863115547500
// MI455X (gfx1250) — matrix operand order repaired and checked by reading, not compiled, not run on hardware
//
#include <hip/hip_runtime.h>
#include <hip/hip_bf16.h>

typedef __attribute__((ext_vector_type(16))) __bf16 v16bf;
typedef __attribute__((ext_vector_type(8)))  __bf16 v8bf;
typedef __attribute__((ext_vector_type(4)))  __bf16 v4bf;
typedef __attribute__((ext_vector_type(8)))  float  v8f;

#define DIM      1152
#define N_HEADS  8
#define HEAD_DIM 144
#define KV_HEADS 2
#define GROUPS   4
#define BSZ      4
#define SEQLEN   2048
#define MROWS    (BSZ*SEQLEN)          /* 8192 */
#define KVN      (2*KV_HEADS*HEAD_DIM) /* 576  */

// ---- bf16 pack helpers (vectorized, branch-free) ----
__device__ __forceinline__ v4bf cvt4(float4 x) {
  v4bf r;
  r[0] = (__bf16)x.x; r[1] = (__bf16)x.y; r[2] = (__bf16)x.z; r[3] = (__bf16)x.w;
  return r;
}
__device__ __forceinline__ v8bf cat8(v4bf a, v4bf b) {
  return __builtin_shufflevector(a, b, 0,1,2,3,4,5,6,7);
}
__device__ __forceinline__ v16bf cat16(v8bf a, v8bf b) {
  return __builtin_shufflevector(a, b, 0,1,2,3,4,5,6,7,8,9,10,11,12,13,14,15);
}

// A-fragment 16x32 (MxK) direct from global (used by flash for Q):
// lane row = lane%16; halves 0..7 = K[8h..], halves 8..15 = K[16+8h..].
__device__ __forceinline__ v16bf load_a_vec(const float* __restrict__ rowptr, int h) {
  const float4* p0 = (const float4*)(rowptr + 8*h);
  const float4* p1 = (const float4*)(rowptr + 16 + 8*h);
  return cat16(cat8(cvt4(p0[0]), cvt4(p0[1])),
               cat8(cvt4(p1[0]), cvt4(p1[1])));
}

// ---- GEMM: C[M,N] = A[M,K] * W[N,K]^T + bias ; K multiple of 32 ----
// 256 threads = 8 waves (4 along M, 2 along N); block tile 128x128.
// Per k-step both operand tiles are converted to bf16 ONCE into LDS;
// fragments are aligned ds_load_b128s, so the wmma pipe sees ~no VALU.
#define GSTRIDE 40   /* 128x32 tile rows padded to 40 bf16 (80 B) */
__global__ __launch_bounds__(256)
void gemm_bias_wmma(const float* __restrict__ A, const float* __restrict__ W,
                    const float* __restrict__ bias, float* __restrict__ C,
                    int M, int N, int K) {
  __shared__ __bf16 Abuf[128][GSTRIDE];
  __shared__ __bf16 Wbuf[128][GSTRIDE];
  const int tid  = threadIdx.x;
  const int lane = tid & 31;
  const int wave = tid >> 5;
  const int h    = lane >> 4;
  const int lr   = lane & 15;
  const int mblk = blockIdx.y * 128;
  const int nblk = blockIdx.x * 128;
  const int m0   = (wave & 3) * 32;      // wave-local tile origin in block
  const int n0   = (wave >> 2) * 64;

  v8f acc[2][4];
#pragma unroll
  for (int i = 0; i < 2; ++i)
#pragma unroll
    for (int j = 0; j < 4; ++j)
#pragma unroll
      for (int r = 0; r < 8; ++r) acc[i][j][r] = 0.0f;

  for (int k0 = 0; k0 < K; k0 += 32) {
    __syncthreads();                     // previous iteration's reads complete
    // cooperative stage: 128 rows x 8 float4 per tensor, 4 per thread each
#pragma unroll
    for (int i = 0; i < 4; ++i) {
      int f  = tid + 256*i;              // 0..1023
      int r  = f >> 3;
      int c4 = f & 7;
      const float4* ap = (const float4*)(A + (size_t)(mblk + r)*K + k0) + c4;
      *(v4bf*)(&Abuf[r][4*c4]) = cvt4(*ap);
      int wn = nblk + r;
      const float4* wp = (const float4*)(W + (size_t)(wn < N ? wn : 0)*K + k0) + c4;
      *(v4bf*)(&Wbuf[r][4*c4]) = cvt4(*wp);
      if (i == 0) {                      // global_prefetch_b8 for next k-step
        __builtin_prefetch(ap + 8, 0, 0);
        __builtin_prefetch(wp + 8, 0, 0);
      }
    }
    __syncthreads();

    v16bf afrag[2];
#pragma unroll
    for (int i = 0; i < 2; ++i) {
      int row = m0 + i*16 + lr;
      v8bf lo = *(const v8bf*)(&Abuf[row][8*h]);
      v8bf hi = *(const v8bf*)(&Abuf[row][16 + 8*h]);
      afrag[i] = cat16(lo, hi);
    }
#pragma unroll
    for (int j = 0; j < 4; ++j) {
      int row = n0 + j*16 + lr;
      v8bf blo = *(const v8bf*)(&Wbuf[row][8*h]);
      v8bf bhi = *(const v8bf*)(&Wbuf[row][16 + 8*h]);
      v16bf bfrag = cat16(blo, bhi);
#pragma unroll
      for (int i = 0; i < 2; ++i)
        acc[i][j] = __builtin_amdgcn_wmma_f32_16x16x32_bf16(
            false, afrag[i], false, bfrag, (short)0, acc[i][j], false, false);
    }
  }

#pragma unroll
  for (int j = 0; j < 4; ++j) {
    int n = nblk + n0 + j*16 + lr;
    if (n < N) {
      float bv = bias[n];
#pragma unroll
      for (int i = 0; i < 2; ++i) {
        int mrow = mblk + m0 + i*16 + 8*h;
#pragma unroll
        for (int r = 0; r < 8; ++r)
          C[(size_t)(mrow + r) * N + n] = acc[i][j][r] + bv;
      }
    }
  }
}

// ---- RoPE on q (in place) and k (in place inside kv buffer) ----
__global__ __launch_bounds__(256)
void rope_kernel(float* __restrict__ qbuf, float* __restrict__ kvbuf,
                 const float* __restrict__ fcos, const float* __restrict__ fsin) {
  const long long QP = (long long)BSZ*SEQLEN*N_HEADS*(HEAD_DIM/2);
  const long long KP = (long long)BSZ*SEQLEN*KV_HEADS*(HEAD_DIM/2);
  long long idx = (long long)blockIdx.x * blockDim.x + threadIdx.x;
  if (idx >= QP + KP) return;
  float* p; int s, d2;
  if (idx < QP) {
    d2 = (int)(idx % (HEAD_DIM/2)); long long t = idx / (HEAD_DIM/2);
    int hh = (int)(t % N_HEADS); t /= N_HEADS;
    s = (int)(t % SEQLEN); int b = (int)(t / SEQLEN);
    p = qbuf + ((size_t)(b*SEQLEN + s)*N_HEADS + hh)*HEAD_DIM + 2*d2;
  } else {
    long long j = idx - QP;
    d2 = (int)(j % (HEAD_DIM/2)); long long t = j / (HEAD_DIM/2);
    int kvh = (int)(t % KV_HEADS); t /= KV_HEADS;
    s = (int)(t % SEQLEN); int b = (int)(t / SEQLEN);
    p = kvbuf + ((size_t)(b*SEQLEN + s)*4 + kvh)*HEAD_DIM + 2*d2;   // slot 0 = k
  }
  float c = fcos[s*(HEAD_DIM/2) + d2], sn = fsin[s*(HEAD_DIM/2) + d2];
  float e = p[0], o = p[1];
  p[0] = e*c - o*sn;
  p[1] = e*sn + o*c;
}

// ---- Flash attention, causal, GQA; K/V tiles staged in LDS as bf16 ----
#define KSTRIDE 160   /* 144 padded, bf16; row = 320 B (32B aligned) */
#define VSTRIDE 48    /* 32 padded,  bf16; row =  96 B (32B aligned) */
__global__ __launch_bounds__(128)
void flash_attn_wmma(const float* __restrict__ q, const float* __restrict__ kv,
                     float* __restrict__ out) {
  __shared__ __bf16 kt[32][KSTRIDE];      // keys: [t][d], d tail zeroed
  __shared__ __bf16 vt[HEAD_DIM][VSTRIDE];// values transposed: [d][t]
  __shared__ __bf16 pbuf[4][16*32];       // per-wave P staging
  const int tid  = threadIdx.x;
  const int lane = tid & 31;
  const int w    = tid >> 5;
  const int h    = lane >> 4;
  const int lr   = lane & 15;
  const int b    = blockIdx.x / N_HEADS;
  const int hh   = blockIdx.x % N_HEADS;
  const int kvh  = hh / GROUPS;
  const int qblk = blockIdx.y * 64;
  const int q0   = qblk + w * 16;
  const float scale = 1.0f / 12.0f;       // HEAD_DIM^-0.5
  const float L2E   = 1.44269504f;

  // Q fragments: 4 full chunks + tail chunk (K 128..143 valid, upper half zero)
  v16bf aq[5];
  {
    const float* qrow = q + ((size_t)(b*SEQLEN + q0 + lr)*N_HEADS + hh)*HEAD_DIM;
#pragma unroll
    for (int c = 0; c < 4; ++c) aq[c] = load_a_vec(qrow + 32*c, h);
    const float4* pt = (const float4*)(qrow + 128 + 8*h);
    v8bf lo = cat8(cvt4(pt[0]), cvt4(pt[1]));
    v8bf zz;
#pragma unroll
    for (int j = 0; j < 8; ++j) zz[j] = (__bf16)0.0f;
    aq[4] = cat16(lo, zz);
  }

  v8f o[9];
#pragma unroll
  for (int c = 0; c < 9; ++c)
#pragma unroll
    for (int r = 0; r < 8; ++r) o[c][r] = 0.0f;
  float rowmax[8], rowsum[8];
#pragma unroll
  for (int r = 0; r < 8; ++r) { rowmax[r] = -3.0e38f; rowsum[r] = 0.0f; }

  for (int t0 = 0; t0 < qblk + 64; t0 += 32) {
    // ---- cooperative staging: 32 rows x 144 floats for K and V ----
#pragma unroll
    for (int i = 0; i < 9; ++i) {
      int f  = tid + 128*i;
      int r  = f / 36;
      int c4 = f % 36;
      size_t rowbase = ((size_t)(b*SEQLEN + t0 + r)*4 + kvh)*HEAD_DIM;
      float4 xk = *(const float4*)(kv + rowbase + 4*c4);
      *(v4bf*)(&kt[r][4*c4]) = cvt4(xk);
      float4 xv = *(const float4*)(kv + rowbase + 2*HEAD_DIM + 4*c4);
      vt[4*c4+0][r] = (__bf16)xv.x;
      vt[4*c4+1][r] = (__bf16)xv.y;
      vt[4*c4+2][r] = (__bf16)xv.z;
      vt[4*c4+3][r] = (__bf16)xv.w;
    }
    if (tid < 32) {                       // zero K tail cols 144..159 (wave-uniform)
      v4bf z;
#pragma unroll
      for (int j = 0; j < 4; ++j) z[j] = (__bf16)0.0f;
#pragma unroll
      for (int j = 0; j < 4; ++j) *(v4bf*)(&kt[tid][144 + 4*j]) = z;
    }
    __syncthreads();

    // ---- S = Q K^T (two 16-key fragments) ----
    v8f s0, s1;
#pragma unroll
    for (int r = 0; r < 8; ++r) { s0[r] = 0.0f; s1[r] = 0.0f; }
#pragma unroll
    for (int c = 0; c < 5; ++c) {
      v16bf bk0 = cat16(*(const v8bf*)(&kt[lr     ][32*c + 8*h]), *(const v8bf*)(&kt[lr     ][32*c + 16 + 8*h]));
      s0 = __builtin_amdgcn_wmma_f32_16x16x32_bf16(false, aq[c], false, bk0,
                                                   (short)0, s0, false, false);
      v16bf bk1 = cat16(*(const v8bf*)(&kt[16 + lr][32*c + 8*h]), *(const v8bf*)(&kt[16 + lr][32*c + 16 + 8*h]));
      s1 = __builtin_amdgcn_wmma_f32_16x16x32_bf16(false, aq[c], false, bk1,
                                                   (short)0, s1, false, false);
    }

    // ---- online softmax (per-row stats replicated across 16-lane halves) ----
    float fac[8];
#pragma unroll
    for (int r = 0; r < 8; ++r) {
      int qrow = q0 + r + 8*h;
      float z0 = s0[r]*scale; if (t0 + lr      > qrow) z0 = -3.0e38f;
      float z1 = s1[r]*scale; if (t0 + 16 + lr > qrow) z1 = -3.0e38f;
      float mx = fmaxf(z0, z1);
      mx = fmaxf(mx, __shfl_xor(mx, 1, 32));
      mx = fmaxf(mx, __shfl_xor(mx, 2, 32));
      mx = fmaxf(mx, __shfl_xor(mx, 4, 32));
      mx = fmaxf(mx, __shfl_xor(mx, 8, 32));
      float nm = fmaxf(rowmax[r], mx);
      float f  = exp2f((rowmax[r] - nm) * L2E);
      float p0 = exp2f((z0 - nm) * L2E);
      float p1 = exp2f((z1 - nm) * L2E);
      float ps = p0 + p1;
      ps += __shfl_xor(ps, 1, 32);
      ps += __shfl_xor(ps, 2, 32);
      ps += __shfl_xor(ps, 4, 32);
      ps += __shfl_xor(ps, 8, 32);
      rowsum[r] = rowsum[r]*f + ps;
      rowmax[r] = nm;
      fac[r] = f;
      pbuf[w][(r + 8*h)*32 + lr]      = (__bf16)p0;
      pbuf[w][(r + 8*h)*32 + 16 + lr] = (__bf16)p1;
    }
    asm volatile("s_wait_dscnt 0" ::: "memory");   // intra-wave LDS round-trip

#pragma unroll
    for (int c = 0; c < 9; ++c)
#pragma unroll
      for (int r = 0; r < 8; ++r)
        o[c][r] = o[c][r] * fac[r];

    // P as A-fragment (16x32) from LDS, aligned vector loads
    v8bf plo = *(const v8bf*)(&pbuf[w][lr*32 + 16*h]);
    v8bf phi = *(const v8bf*)(&pbuf[w][lr*32 + 16*h + 8]);
    v16bf ap = cat16(plo, phi);

    // ---- O += P V : 9 chunks of 16 along HEAD_DIM, V^T rows contiguous ----
#pragma unroll
    for (int c = 0; c < 9; ++c) {
      v16bf bv = *(const v16bf*)(&vt[16*c + lr][16*h]);
      o[c] = __builtin_amdgcn_wmma_f32_16x16x32_bf16(false, ap, false, bv,
                                                     (short)0, o[c], false, false);
    }
    __syncthreads();   // all reads done before next tile overwrites LDS
  }

  // epilogue: normalize and store (b,s,h,d)
#pragma unroll
  for (int r = 0; r < 8; ++r) {
    float inv = 1.0f / rowsum[r];
    int m = q0 + r + 8*h;
    float* orow = out + ((size_t)(b*SEQLEN + m)*N_HEADS + hh)*HEAD_DIM;
#pragma unroll
    for (int c = 0; c < 9; ++c)
      orow[16*c + lr] = o[c][r] * inv;
  }
}

extern "C" void kernel_launch(void* const* d_in, const int* in_sizes, int n_in,
                              void* d_out, int out_size, void* d_ws, size_t ws_size,
                              hipStream_t stream) {
  const float* x   = (const float*)d_in[0];
  const float* wq  = (const float*)d_in[1];
  const float* bq  = (const float*)d_in[2];
  const float* wkv = (const float*)d_in[3];
  const float* bkv = (const float*)d_in[4];
  const float* wo  = (const float*)d_in[5];
  const float* bo  = (const float*)d_in[6];
  const float* fc  = (const float*)d_in[7];
  const float* fs  = (const float*)d_in[8];
  float* out = (float*)d_out;

  float* q_ws  = (float*)d_ws;                       // 8192*1152
  float* kv_ws = q_ws + (size_t)MROWS*DIM;           // 8192*576 (k,v interleaved)
  float* ao_ws = kv_ws + (size_t)MROWS*KVN;          // 8192*1152

  dim3 blk(256);
  gemm_bias_wmma<<<dim3(DIM/128, MROWS/128), blk, 0, stream>>>(x, wq, bq, q_ws,
                                                               MROWS, DIM, DIM);
  gemm_bias_wmma<<<dim3((KVN + 127)/128, MROWS/128), blk, 0, stream>>>(x, wkv, bkv, kv_ws,
                                                                       MROWS, KVN, DIM);
  {
    long long total = (long long)BSZ*SEQLEN*(HEAD_DIM/2)*(N_HEADS + KV_HEADS);
    int grid = (int)((total + 255) / 256);
    rope_kernel<<<grid, 256, 0, stream>>>(q_ws, kv_ws, fc, fs);
  }
  flash_attn_wmma<<<dim3(BSZ*N_HEADS, SEQLEN/64), dim3(128), 0, stream>>>(q_ws, kv_ws, ao_ws);
  gemm_bias_wmma<<<dim3(DIM/128, MROWS/128), blk, 0, stream>>>(ao_ws, wo, bo, out,
                                                               MROWS, DIM, DIM);
}
